// Encoder_91036126806046
// MI455X (gfx1250) — compile-verified, matrix operand order checked by reading, not run on hardware
//
#include <hip/hip_runtime.h>
#include <hip/hip_bf16.h>
#include <math.h>

// Problem constants (from reference)
#define VOCAB 10000
#define EMB   300
#define EMBP  320      // K padded to multiple of 32 for wmma_f32_16x16x32_bf16
#define UNITS 512
#define GATES 2048     // 4*UNITS
#define BATCH 128
#define SEQ   1024
#define BT    (BATCH*SEQ)

typedef __bf16 bf16_t;
typedef __attribute__((ext_vector_type(8)))  bf16_t v8bf;
typedef __attribute__((ext_vector_type(16))) bf16_t v16bf;
typedef __attribute__((ext_vector_type(8)))  float  v8f;

// ---------------------------------------------------------------------------
// WMMA helpers (CDNA5: V_WMMA_F32_16X16X32_BF16, wave32)
// ---------------------------------------------------------------------------
__device__ __forceinline__ v8f wmma_bf16(v16bf a, v16bf b, v8f c) {
  // 8 args: (neg_a, A, neg_b, B, c_mod, C, reuse_a, reuse_b)
  return __builtin_amdgcn_wmma_f32_16x16x32_bf16(false, a, false, b,
                                                 (short)0, c, false, false);
}

// A-matrix 16x32 bf16 fragment (ISA 7.12.2): lane L (L<16) holds row M=L,
// K in {h8..h8+7} U {16+h8..16+h8+7} with h8 = 8*(L>>4). From a row-major
// bf16 row pointer that is two contiguous 16-byte chunks.
__device__ __forceinline__ v16bf load_a_frag(const bf16_t* __restrict__ row,
                                             int kc, int half) {
  const v8bf lo = *(const v8bf*)(row + kc * 32 + half * 8);
  const v8bf hi = *(const v8bf*)(row + kc * 32 + 16 + half * 8);
  v16bf a;
#pragma unroll
  for (int j = 0; j < 8; ++j) { a[j] = lo[j]; a[8 + j] = hi[j]; }
  return a;
}

// ---------------------------------------------------------------------------
// One-time prep kernels
// ---------------------------------------------------------------------------

// Pre-swizzle a (Krows x 2048) fp32 matrix into per-lane B-fragment order:
// dst[((ntile*KC + kc)*32 + lane)*16 + j], so each lane's 16 bf16 B elements
// are one contiguous 32B chunk (two coalesced b128 loads at use time).
__global__ __launch_bounds__(256) void swizzle_b_kernel(
    const float* __restrict__ src, bf16_t* __restrict__ dst,
    int KC, int Krows, int total) {
  int idx = blockIdx.x * 256 + threadIdx.x;
  if (idx >= total) return;
  int j     = idx & 15;
  int rest  = idx >> 4;
  int lane  = rest & 31;
  rest    >>= 5;
  int kc    = rest % KC;
  int ntile = rest / KC;
  int halfv = lane >> 4;
  int kk    = (j < 8) ? (halfv * 8 + j) : (16 + halfv * 8 + (j - 8));
  int k     = kc * 32 + kk;
  int col   = ntile * 16 + (lane & 15);
  float v   = (k < Krows) ? src[(size_t)k * GATES + col] : 0.0f;
  dst[idx] = (bf16_t)v;
}

// Embedding gather + fp32->bf16, K padded 300->320 with zeros.
__global__ __launch_bounds__(256) void gather_embed_kernel(
    const int* __restrict__ tokens, const float* __restrict__ emb,
    bf16_t* __restrict__ Xb) {
  size_t idx = (size_t)blockIdx.x * 256 + threadIdx.x;
  if (idx >= (size_t)BT * EMBP) return;
  int    e  = (int)(idx % EMBP);
  size_t bt = idx / EMBP;
  float  v  = 0.0f;
  if (e < EMB) {
    int tok = tokens[bt];
    v = emb[(size_t)tok * EMB + e];
  }
  Xb[idx] = (bf16_t)v;
}

__global__ __launch_bounds__(256) void init_state_kernel(
    const float* __restrict__ h0, const float* __restrict__ c0,
    float* __restrict__ h32, bf16_t* __restrict__ hbf,
    float* __restrict__ c32) {
  int idx = blockIdx.x * 256 + threadIdx.x;
  if (idx >= BATCH * UNITS) return;
  float h = h0[idx], c = c0[idx];
  h32[idx] = h;
  hbf[idx] = (bf16_t)h;
  c32[idx] = c;
}

// ---------------------------------------------------------------------------
// zx = Xb(131072x320)bf16 @ W(320x2048)bf16 + b  -> fp32 (161 GFLOP, WMMA)
// Block = 8 waves; wave owns one 16x16 output tile (A reused via L2/WGP$).
// ---------------------------------------------------------------------------
__global__ __launch_bounds__(256) void zx_gemm_kernel(
    const bf16_t* __restrict__ Xb, const bf16_t* __restrict__ Wfrag,
    const float* __restrict__ bias, float* __restrict__ zx) {
  const int lane  = threadIdx.x & 31;
  const int wave  = threadIdx.x >> 5;
  const int ntile = blockIdx.x * 8 + wave;   // 0..127
  const int mtile = blockIdx.y;              // 0..8191
  const int half  = lane >> 4;
  const bf16_t* arow  = Xb + (size_t)(mtile * 16 + (lane & 15)) * EMBP;
  const bf16_t* bbase = Wfrag + (size_t)ntile * 10 * 32 * 16;

  v8f acc = {};
#pragma unroll
  for (int kc = 0; kc < 10; ++kc) {
    if (kc < 9) {  // hint next A chunk into cache (global_prefetch_b8)
      __builtin_prefetch(arow + (kc + 1) * 32, 0, 0);
    }
    v16bf a = load_a_frag(arow, kc, half);
    v16bf b = *(const v16bf*)(bbase + ((size_t)kc * 32 + lane) * 16);
    acc = wmma_bf16(a, b, acc);
  }

  // C/D layout: lane holds N = lane&15, rows M = v + 8*(lane>>4)
  const int   col = ntile * 16 + (lane & 15);
  const float bv  = bias[col];
#pragma unroll
  for (int v = 0; v < 8; ++v) {
    int r = mtile * 16 + v + 8 * half;
    zx[(size_t)r * GATES + col] = acc[v] + bv;
  }
}

// ---------------------------------------------------------------------------
// One LSTM time step: z = zx[:,t,:] + h@U ; gates ; masked carry update.
// grid(32 unit-tiles, 8 batch-tiles), 1 wave per block.
// 4 gate accumulators per tile; K=512 -> 16 k-chunks -> 64 WMMAs per wave.
// ---------------------------------------------------------------------------
__global__ __launch_bounds__(32) void lstm_step_kernel(
    const float* __restrict__ zx, const bf16_t* __restrict__ Ufrag,
    const int* __restrict__ tokens,
    const float* __restrict__ h_in32, const bf16_t* __restrict__ h_inbf,
    const float* __restrict__ c_in,
    float* __restrict__ h_out32, bf16_t* __restrict__ h_outbf,
    float* __restrict__ c_out, float* __restrict__ enc_out, int t) {
  const int lane = threadIdx.x;
  const int n    = blockIdx.x;   // unit tile 0..31
  const int m    = blockIdx.y;   // batch tile 0..7
  const int half = lane >> 4;
  const bf16_t* hrow = h_inbf + (size_t)(m * 16 + (lane & 15)) * UNITS;

  v8f acc0 = {}, acc1 = {}, acc2 = {}, acc3 = {};
#pragma unroll 4
  for (int kc = 0; kc < 16; ++kc) {
    v16bf a = load_a_frag(hrow, kc, half);
    // gate g lives in column tiles n + g*32 of the 2048-wide U
    const v16bf b0 = *(const v16bf*)(Ufrag + (((size_t)(n      ) * 16 + kc) * 32 + lane) * 16);
    const v16bf b1 = *(const v16bf*)(Ufrag + (((size_t)(n +  32) * 16 + kc) * 32 + lane) * 16);
    const v16bf b2 = *(const v16bf*)(Ufrag + (((size_t)(n +  64) * 16 + kc) * 32 + lane) * 16);
    const v16bf b3 = *(const v16bf*)(Ufrag + (((size_t)(n +  96) * 16 + kc) * 32 + lane) * 16);
    acc0 = wmma_bf16(a, b0, acc0);
    acc1 = wmma_bf16(a, b1, acc1);
    acc2 = wmma_bf16(a, b2, acc2);
    acc3 = wmma_bf16(a, b3, acc3);
  }

  const int ucol = n * 16 + (lane & 15);
#pragma unroll
  for (int v = 0; v < 8; ++v) {
    const int    bidx = m * 16 + v + 8 * half;
    const size_t zrow = ((size_t)bidx * SEQ + t) * GATES;
    float zi = acc0[v] + zx[zrow + 0 * UNITS + ucol];
    float zf = acc1[v] + zx[zrow + 1 * UNITS + ucol];
    float zg = acc2[v] + zx[zrow + 2 * UNITS + ucol];
    float zo = acc3[v] + zx[zrow + 3 * UNITS + ucol];
    float ig = 1.0f / (1.0f + expf(-zi));
    float fg = 1.0f / (1.0f + expf(-zf));
    float gg = tanhf(zg);
    float og = 1.0f / (1.0f + expf(-zo));
    float cold = c_in[(size_t)bidx * UNITS + ucol];
    float cnew = fg * cold + ig * gg;
    float hnew = og * tanhf(cnew);
    bool  msk  = tokens[(size_t)bidx * SEQ + t] != 0;
    float hold = h_in32[(size_t)bidx * UNITS + ucol];
    float hv = msk ? hnew : hold;
    float cv = msk ? cnew : cold;
    h_out32[(size_t)bidx * UNITS + ucol] = hv;
    h_outbf[(size_t)bidx * UNITS + ucol] = (bf16_t)hv;
    c_out  [(size_t)bidx * UNITS + ucol] = cv;
    enc_out[((size_t)bidx * SEQ + t) * UNITS + ucol] = hv;
  }
}

__global__ __launch_bounds__(256) void finalize_kernel(
    const float* __restrict__ h, const float* __restrict__ c,
    float* __restrict__ out) {
  int idx = blockIdx.x * 256 + threadIdx.x;
  if (idx >= BATCH * UNITS) return;
  out[idx] = h[idx];
  out[BATCH * UNITS + idx] = c[idx];
}

// ---------------------------------------------------------------------------
// Workspace layout (bytes, 256-aligned)
// ---------------------------------------------------------------------------
static const size_t XB_OFF    = 0;                                  // 80 MiB bf16
static const size_t WFRAG_OFF = XB_OFF    + (size_t)BT * EMBP * 2;  // 1.25 MiB
static const size_t UFRAG_OFF = WFRAG_OFF + (size_t)EMBP * GATES * 2;
static const size_t ZX_OFF    = UFRAG_OFF + (size_t)UNITS * GATES * 2;  // 1 GiB fp32
static const size_t H32_OFF   = ZX_OFF    + (size_t)BT * GATES * 4; // 2x ping-pong
static const size_t HBF_OFF   = H32_OFF   + (size_t)2 * BATCH * UNITS * 4;
static const size_t C32_OFF   = HBF_OFF   + (size_t)2 * BATCH * UNITS * 2;

extern "C" void kernel_launch(void* const* d_in, const int* in_sizes, int n_in,
                              void* d_out, int out_size, void* d_ws, size_t ws_size,
                              hipStream_t stream) {
  const int*   tokens = (const int*)  d_in[0];  // (B,T) int32
  const float* h0     = (const float*)d_in[1];  // (B,U)
  const float* c0     = (const float*)d_in[2];  // (B,U)
  const float* emb    = (const float*)d_in[3];  // (V,E)
  const float* W      = (const float*)d_in[4];  // (E,4U)
  const float* U      = (const float*)d_in[5];  // (U,4U)
  const float* bias   = (const float*)d_in[6];  // (4U,)
  float* out = (float*)d_out;                   // enc(B,T,U) ++ h1 ++ c1

  char* ws = (char*)d_ws;
  bf16_t* Xb    = (bf16_t*)(ws + XB_OFF);
  bf16_t* Wfrag = (bf16_t*)(ws + WFRAG_OFF);
  bf16_t* Ufrag = (bf16_t*)(ws + UFRAG_OFF);
  float*  zx    = (float*) (ws + ZX_OFF);
  float*  h32   = (float*) (ws + H32_OFF);   // [2][B*U]
  bf16_t* hbf   = (bf16_t*)(ws + HBF_OFF);   // [2][B*U]
  float*  c32   = (float*) (ws + C32_OFF);   // [2][B*U]

  // 1) One-time weight swizzles into WMMA B-fragment order (bf16)
  {
    int totW = 128 * 10 * 32 * 16;   // ntiles * KC(320/32) * 32 lanes * 16
    swizzle_b_kernel<<<(totW + 255) / 256, 256, 0, stream>>>(W, Wfrag, 10, EMB, totW);
    int totU = 128 * 16 * 32 * 16;   // KC = 512/32
    swizzle_b_kernel<<<(totU + 255) / 256, 256, 0, stream>>>(U, Ufrag, 16, UNITS, totU);
  }

  // 2) Embedding gather -> bf16, padded
  {
    size_t tot = (size_t)BT * EMBP;
    gather_embed_kernel<<<(int)((tot + 255) / 256), 256, 0, stream>>>(tokens, emb, Xb);
  }

  // 3) Init ping-pong state from h0/c0
  init_state_kernel<<<(BATCH * UNITS + 255) / 256, 256, 0, stream>>>(
      h0, c0, h32, hbf, c32);

  // 4) Big input-projection GEMM: zx = Xb @ W + b  (WMMA bf16 -> f32)
  {
    dim3 grid(16, BT / 16);  // 16 blocks of 8 waves cover 128 n-tiles; 8192 m-tiles
    zx_gemm_kernel<<<grid, 256, 0, stream>>>(Xb, Wfrag, bias, zx);
  }

  // 5) Sequential scan: 1024 step kernels with ping-pong h/c
  for (int t = 0; t < SEQ; ++t) {
    const int pin  = t & 1;
    const int pout = 1 - pin;
    dim3 grid(UNITS / 16, BATCH / 16);  // (32, 8), 1 wave each
    lstm_step_kernel<<<grid, 32, 0, stream>>>(
        zx, Ufrag, tokens,
        h32 + (size_t)pin  * BATCH * UNITS,
        hbf + (size_t)pin  * BATCH * UNITS,
        c32 + (size_t)pin  * BATCH * UNITS,
        h32 + (size_t)pout * BATCH * UNITS,
        hbf + (size_t)pout * BATCH * UNITS,
        c32 + (size_t)pout * BATCH * UNITS,
        out, t);
  }

  // 6) Append h1, c1 (after 1024 steps, state lives in buffer 0)
  finalize_kernel<<<(BATCH * UNITS + 255) / 256, 256, 0, stream>>>(
      h32, c32, out + (size_t)BT * UNITS);
}
